// LocalCausalGraph_4767413698783
// MI455X (gfx1250) — matrix operand order repaired and checked by reading, not compiled, not run on hardware
//
#include <hip/hip_runtime.h>
#include <hip/hip_bf16.h>

#define D_MODEL 1024
#define CD 64
#define BB 4
#define LL 512
#define M_TOT (BB * LL)   // 2048
#define N_TOT (2 * CD)    // 128

typedef __attribute__((ext_vector_type(16))) __bf16 v16bf;
typedef __attribute__((ext_vector_type(8)))  float  v8f;

union FragBF { uint4 u[2]; v16bf v; };

__device__ __forceinline__ unsigned short f2bf(float f) {
    unsigned int u = __float_as_uint(f);
    unsigned int r = u + 0x7FFFu + ((u >> 16) & 1u);   // round-to-nearest-even
    return (unsigned short)(r >> 16);
}

// ---------------- prep 1: x (f32) -> xb (bf16) ----------------
__global__ void cvt_x_kernel(const float* __restrict__ x, unsigned short* __restrict__ xb) {
    int i = (blockIdx.x * 256 + threadIdx.x) * 4;   // total 2048*1024, grid covers exactly
    float4 f = *(const float4*)(x + i);
    ushort4 o;
    o.x = f2bf(f.x); o.y = f2bf(f.y); o.z = f2bf(f.z); o.w = f2bf(f.w);
    *(ushort4*)(xb + i) = o;
}

// ---------------- prep 2: fused weight C = [W1[:, :64]@Wc ; W1[:, 64:]@We] (128x1024, bf16) ----
__global__ void build_c_kernel(const float* __restrict__ Wc, const float* __restrict__ We,
                               const float* __restrict__ W1, unsigned short* __restrict__ cb) {
    int idx = blockIdx.x * 256 + threadIdx.x;       // 0 .. 128*1024-1
    int n = idx >> 10;
    int d = idx & 1023;
    const float* Wr = (n < CD) ? Wc : We;
    const float* w1row = W1 + (n & (CD - 1)) * (2 * CD) + ((n < CD) ? 0 : CD);
    float s = 0.0f;
#pragma unroll 8
    for (int c = 0; c < CD; ++c) s += w1row[c] * Wr[c * D_MODEL + d];
    cb[idx] = f2bf(s);
}

// ---------------- main GEMM: G(2048x128) = xb(2048x1024) @ cb^T, +b1 folded into hc half ------
__global__ void gemm_wmma_kernel(const unsigned short* __restrict__ xb,
                                 const unsigned short* __restrict__ cb,
                                 const float* __restrict__ b1,
                                 float* __restrict__ G) {
    const int lane = threadIdx.x & 31;
    const int wave = threadIdx.x >> 5;       // 0..7 -> N tile
    const int mTile = blockIdx.x << 4;       // 16 rows per block
    const int nTile = wave << 4;

    const int half = lane >> 4;              // 0 or 1
    const int lid  = lane & 15;

    const uint4* arow = (const uint4*)(xb + (size_t)(mTile + lid) * D_MODEL);
    const uint4* brow = (const uint4*)(cb + (size_t)(nTile + lid) * D_MODEL);

    v8f acc = {};
#pragma unroll 4
    for (int k0 = 0; k0 < D_MODEL; k0 += 32) {
        FragBF a, b;
        // A: row m = lid per lane; lanes 0-15 hold K = k0..k0+15
        //                          lanes 16-31: K = k0+16..k0+31
        int ak = (k0 + half * 16) >> 3;      // uint4 = 8 bf16
        a.u[0] = arow[ak];
        a.u[1] = arow[ak + 1];
        // B: column n = lid per lane; lanes 0-15 hold K=k0..k0+15, lanes 16-31 K=k0+16..k0+31
        int bk = (k0 + half * 16) >> 3;
        b.u[0] = brow[bk];
        b.u[1] = brow[bk + 1];
        acc = __builtin_amdgcn_wmma_f32_16x16x32_bf16(
            false, a.v, false, b.v, (short)0, acc, false, false);
    }

    const int n = nTile + lid;
    const float bias = (n < CD) ? b1[n] : 0.0f;   // fold b1 into hc half only
    const int mBase = mTile + half * 8;
    float* gout = G + (size_t)mBase * N_TOT + n;
#pragma unroll
    for (int r = 0; r < 8; ++r) gout[(size_t)r * N_TOT] = acc[r] + bias;
}

// ---------------- pairwise: out[b,i,j] = sigmoid(b2 + sum_c gelu(hc[i,c]+he[j,c]) * w2[c]) ----
__global__ void pairwise_kernel(const float* __restrict__ G,
                                const float* __restrict__ W2,
                                const float* __restrict__ b2,
                                float* __restrict__ out) {
    __shared__ float shc[32][65];   // padded: kill 64-bank conflicts
    __shared__ float she[32][65];
    __shared__ float sw2[CD];

    const int b     = blockIdx.z;
    const int iTile = blockIdx.y * 32;
    const int jTile = blockIdx.x * 32;
    const int tid   = threadIdx.x;  // 256

    for (int t = tid; t < 32 * CD; t += 256) {
        int r = t >> 6, c = t & (CD - 1);
        shc[r][c] = G[(size_t)((b * LL + iTile + r) * N_TOT) + c];
        she[r][c] = G[(size_t)((b * LL + jTile + r) * N_TOT) + CD + c];
    }
    if (tid < CD) sw2[tid] = W2[tid];
    __syncthreads();

    const float bias2 = b2[0];
    const int il = tid >> 3;          // 0..31
    const int jl = (tid & 7) * 4;     // 0,4,...,28

    float s0 = 0.f, s1 = 0.f, s2 = 0.f, s3 = 0.f;
    const float ir2 = 0.70710678118654752f;
#pragma unroll 4
    for (int c = 0; c < CD; ++c) {
        float hv = shc[il][c];
        float w  = sw2[c];
        float v0 = hv + she[jl + 0][c];
        float v1 = hv + she[jl + 1][c];
        float v2 = hv + she[jl + 2][c];
        float v3 = hv + she[jl + 3][c];
        // exact GELU: 0.5*v*(1+erf(v/sqrt(2)))
        s0 += w * (0.5f * v0 * (1.0f + erff(v0 * ir2)));
        s1 += w * (0.5f * v1 * (1.0f + erff(v1 * ir2)));
        s2 += w * (0.5f * v2 * (1.0f + erff(v2 * ir2)));
        s3 += w * (0.5f * v3 * (1.0f + erff(v3 * ir2)));
    }

    float* o = out + (size_t)(b * LL + iTile + il) * LL + jTile + jl;
    o[0] = 1.0f / (1.0f + expf(-(s0 + bias2)));
    o[1] = 1.0f / (1.0f + expf(-(s1 + bias2)));
    o[2] = 1.0f / (1.0f + expf(-(s2 + bias2)));
    o[3] = 1.0f / (1.0f + expf(-(s3 + bias2)));
}

extern "C" void kernel_launch(void* const* d_in, const int* in_sizes, int n_in,
                              void* d_out, int out_size, void* d_ws, size_t ws_size,
                              hipStream_t stream) {
    const float* x  = (const float*)d_in[0];
    const float* Wc = (const float*)d_in[1];
    const float* We = (const float*)d_in[2];
    const float* W1 = (const float*)d_in[3];
    const float* b1 = (const float*)d_in[4];
    const float* W2 = (const float*)d_in[5];
    const float* b2 = (const float*)d_in[6];
    float* out = (float*)d_out;

    char* ws = (char*)d_ws;
    unsigned short* xb = (unsigned short*)ws;                                   // 4 MB
    unsigned short* cb = (unsigned short*)(ws + (size_t)4 * 1024 * 1024);       // 256 KB
    float*          G  = (float*)(ws + (size_t)4 * 1024 * 1024 + 256 * 1024);   // 1 MB

    // 1) x -> bf16  (2048*1024 elems, 4 per thread)
    cvt_x_kernel<<<(M_TOT * D_MODEL) / (256 * 4), 256, 0, stream>>>(x, xb);
    // 2) fused weight C (128*1024 elems, 1 per thread)
    build_c_kernel<<<(N_TOT * D_MODEL) / 256, 256, 0, stream>>>(Wc, We, W1, cb);
    // 3) WMMA GEMM: G = xb @ cb^T (+b1 on hc half)
    gemm_wmma_kernel<<<M_TOT / 16, 256, 0, stream>>>(xb, cb, b1, G);
    // 4) pairwise GELU + dot + sigmoid
    pairwise_kernel<<<dim3(LL / 32, LL / 32, BB), 256, 0, stream>>>(G, W2, b2, out);
}
